// Multi_Head_Self_Attention_70987219468562
// MI455X (gfx1250) — matrix operand order repaired and checked by reading, not compiled, not run on hardware
//
#include <hip/hip_runtime.h>
#include <hip/hip_bf16.h>

// MI455X (gfx1250) implementation.
// Big GEMMs (edge@Wk^T, edge_hidden@We^T) use v_wmma_f32_16x16x32_f16 with
// f32 accumulation. A-tiles are staged once per block in LDS (f16) and the
// per-wave WMMA fragments are pulled via ds_load_b128; B fragments stream
// from the L0/L2-resident 128 KB f16 weight matrices. Attention core stays
// fp32 VALU (<2% of work).

typedef __attribute__((ext_vector_type(16))) _Float16 v16h;
typedef __attribute__((ext_vector_type(8)))  _Float16 v8h;
typedef __attribute__((ext_vector_type(8)))  float    v8f;

#define B_   8
#define M_   128
#define HID_ 256
#define NH_  8
#define D_   32
#define R_EDGE (B_*M_*M_)     /* 131072 rows of the edge GEMMs */
#define LDSP (HID_ + 8)       /* padded LDS row stride (halves): 132 dwords */

__device__ __forceinline__ float mish_f(float x) {
    float sp = (x > 20.f) ? x : log1pf(expf(x));
    return x * tanhf(sp);
}

// ---------------- kernel 0: weights f32 -> f16 --------------------------
__global__ void cvt_weights_k(const float* __restrict__ Wk, const float* __restrict__ We,
                              _Float16* __restrict__ Wk16, _Float16* __restrict__ We16) {
    int i = blockIdx.x * blockDim.x + threadIdx.x;
    if (i < HID_ * HID_)            Wk16[i] = (_Float16)Wk[i];
    else if (i < 2 * HID_ * HID_)   We16[i - HID_ * HID_] = (_Float16)We[i - HID_ * HID_];
}

// ---------------- kernel 1: q/v projections (fp32, tiny) ----------------
__global__ void qv_proj_k(const float* __restrict__ node,
                          const float* __restrict__ Wq, const float* __restrict__ bq,
                          const float* __restrict__ Wv, const float* __restrict__ bv,
                          float* __restrict__ q, float* __restrict__ v) {
    int i = blockIdx.x * blockDim.x + threadIdx.x;     // [0, 2*B*M*HID)
    const int half = B_ * M_ * HID_;
    bool isV = i >= half;
    int idx = isV ? i - half : i;
    int o  = idx & (HID_ - 1);
    int rm = idx >> 8;                                 // b*M + m
    const float* x = node + (long)rm * HID_;
    const float* W = (isV ? Wv : Wq) + (long)o * HID_;
    float acc = (isV ? bv : bq)[o];
    for (int c = 0; c < HID_; ++c) acc += x[c] * W[c];
    (isV ? v : q)[idx] = acc;
}

// ---------------- kernel 2: k = edge @ Wk^T + bk  (WMMA) ----------------
// block = 128 thr = 4 waves; block covers 16 rows x 256 cols.
// Stage A tile (16x256 f16) in LDS once; wave w covers cols [64w, 64w+64)
// as 4 16x16 tiles; all 8 A fragments held in registers; 32 wmma per wave.
__global__ __launch_bounds__(128) void kproj_wmma_k(const float* __restrict__ edge,
        const _Float16* __restrict__ Wk16, const float* __restrict__ bk,
        _Float16* __restrict__ k16) {
    __shared__ _Float16 As[16 * LDSP];
    const int tid = threadIdx.x;
    const long r0 = (long)blockIdx.x * 16;

    {   // stage + convert: thread -> (row = tid>>3, 32-col chunk = tid&7)
        int row = tid >> 3;
        int c0  = (tid & 7) * 32;
        const float* src = edge + (r0 + row) * HID_ + c0;
        _Float16* dst = As + row * LDSP + c0;
        #pragma unroll
        for (int i = 0; i < 32; ++i) dst[i] = (_Float16)src[i];
    }
    __syncthreads();

    const int lane = tid & 31;
    const int wv   = tid >> 5;
    const int lrow = lane & 15;
    const int kb   = (lane < 16) ? 0 : 16;   // A-fragment K base (contiguous K, matches B)
    const int boff = (lane < 16) ? 0 : 16;   // B-fragment K base (contiguous K, matches A)

    v16h A[8];
    #pragma unroll
    for (int ks = 0; ks < 8; ++ks) {
        const _Float16* p = As + lrow * LDSP + ks * 32 + kb;
        v8h lo = *reinterpret_cast<const v8h*>(p);
        v8h hi = *reinterpret_cast<const v8h*>(p + 8);
        A[ks] = __builtin_shufflevector(lo, hi, 0,1,2,3,4,5,6,7,8,9,10,11,12,13,14,15);
    }

    v8f z; for (int i = 0; i < 8; ++i) z[i] = 0.f;
    v8f acc[4]; for (int t = 0; t < 4; ++t) acc[t] = z;

    #pragma unroll
    for (int ks = 0; ks < 8; ++ks) {
        #pragma unroll
        for (int t = 0; t < 4; ++t) {
            int o = 64 * wv + 16 * t + lrow;
            v16h bf = *reinterpret_cast<const v16h*>(Wk16 + (long)o * HID_ + ks * 32 + boff);
            acc[t] = __builtin_amdgcn_wmma_f32_16x16x32_f16(false, A[ks], false, bf,
                                                            (short)0, acc[t], false, false);
        }
    }

    const int rowhi = (lane < 16) ? 0 : 8;
    #pragma unroll
    for (int t = 0; t < 4; ++t) {
        int o = 64 * wv + 16 * t + lrow;
        float bias = bk[o];
        #pragma unroll
        for (int g = 0; g < 8; ++g)
            k16[(r0 + g + rowhi) * HID_ + o] = (_Float16)(acc[t][g] + bias);
    }
}

// ---------------- kernel 3a: scores s[b,h,m,n] = scale * q.k ------------
__global__ void scores_k(const float* __restrict__ q, const _Float16* __restrict__ k16,
                         float* __restrict__ s) {
    int r = blockIdx.x * blockDim.x + threadIdx.x;     // (b,m,n)
    if (r >= R_EDGE) return;
    int b = r >> 14, rem = r & 16383, m = rem >> 7, n = rem & 127;
    const float* qrow = q + ((long)b * M_ + m) * HID_;
    const _Float16* krow = k16 + (long)r * HID_;
    const float scale = 0.17677669529663689f;          // 1/sqrt(32)
    for (int h = 0; h < NH_; ++h) {
        float acc = 0.f;
        #pragma unroll
        for (int d = 0; d < D_; ++d) acc += qrow[h * D_ + d] * (float)krow[h * D_ + d];
        s[(((long)b * NH_ + h) * M_ + m) * M_ + n] = acc * scale;
    }
}

// ---------------- kernel 3b: masked row & column softmax stats ----------
__global__ void stats_k(const float* __restrict__ s, const float* __restrict__ mask,
                        float* __restrict__ rowmax, float* __restrict__ rowsum,
                        float* __restrict__ colmax, float* __restrict__ colsum) {
    int i = blockIdx.x * blockDim.x + threadIdx.x;     // (b,h)*M + idx
    if (i >= B_ * NH_ * M_) return;
    int bh  = i >> 7;
    int idx = i & (M_ - 1);
    int b   = bh >> 3;
    const float* mrow = mask + b * M_;
    long base = (long)bh * (M_ * M_);
    float rmx = -1e30f, cmx = -1e30f;
    for (int n = 0; n < M_; ++n) {
        if (mrow[n] != 0.f) {
            rmx = fmaxf(rmx, s[base + (long)idx * M_ + n]);   // row idx
            cmx = fmaxf(cmx, s[base + (long)n * M_ + idx]);   // col idx (mask on 1st index)
        }
    }
    float rs = 0.f, cs = 0.f;
    for (int n = 0; n < M_; ++n) {
        if (mrow[n] != 0.f) {
            rs += expf(s[base + (long)idx * M_ + n] - rmx);
            cs += expf(s[base + (long)n * M_ + idx] - cmx);
        }
    }
    rowmax[i] = rmx; rowsum[i] = rs; colmax[i] = cmx; colsum[i] = cs;
}

// ---------------- kernel 3c: message = (out_attn + in_attn - diag)*dm ---
// in_scores[b,h,i,j] == s[b,h,j,i]; both softmaxes zero out masked n.
__global__ void message_k(const float* __restrict__ s, const float* __restrict__ dist,
        const float* __restrict__ mask, const float* __restrict__ lam,
        const float* __restrict__ rowmax, const float* __restrict__ rowsum,
        const float* __restrict__ colmax, const float* __restrict__ colsum,
        float* __restrict__ msg) {
    long i = (long)blockIdx.x * blockDim.x + threadIdx.x;   // (b,h,m,n)
    if (i >= (long)B_ * NH_ * M_ * M_) return;
    int n = (int)(i & 127); long t = i >> 7;
    int m = (int)(t & 127); long bh = t >> 7;
    int b = (int)(bh >> 3);
    float out = 0.f;
    if (mask[b * M_ + n] != 0.f) {
        long base = bh * (M_ * M_);
        float sm = s[base + (long)m * M_ + n];
        float st = s[base + (long)n * M_ + m];
        long ri = bh * M_ + m;
        float oa = expf(sm - rowmax[ri]) / rowsum[ri];        // out_attn[m,n]
        float ia = expf(st - colmax[ri]) / colsum[ri];        // in_attn[m,n]
        float mg = (m == n) ? ia : (oa + ia);                 // minus diag at m==n
        float dm = expf(-lam[0] * dist[((long)b * M_ + m) * M_ + n]);
        out = mg * dm;
    }
    msg[i] = out;
}

// ---------------- kernel 4a: node_hidden = msg @ v ----------------------
__global__ void nodehid_k(const float* __restrict__ msg, const float* __restrict__ v,
                          float* __restrict__ nh) {
    int i = blockIdx.x * blockDim.x + threadIdx.x;     // (b,m,hid)
    if (i >= B_ * M_ * HID_) return;
    int c = i & (HID_ - 1); int bm = i >> 8; int m = bm & 127; int b = bm >> 7;
    int h = c >> 5;
    const float* mrow = msg + (((long)b * NH_ + h) * M_ + m) * M_;
    const float* vcol = v + (long)b * M_ * HID_ + c;
    float acc = 0.f;
    for (int n = 0; n < M_; ++n) acc += mrow[n] * vcol[(long)n * HID_];
    nh[i] = acc;
}

// ---------------- kernel 4b: node out = mish(nh @ Wn^T + bn) ------------
__global__ void nodeout_k(const float* __restrict__ nh, const float* __restrict__ Wn,
                          const float* __restrict__ bn, float* __restrict__ outn) {
    int i = blockIdx.x * blockDim.x + threadIdx.x;     // (b*m, o)
    if (i >= B_ * M_ * HID_) return;
    int o = i & (HID_ - 1); int bm = i >> 8;
    const float* x = nh + (long)bm * HID_;
    const float* w = Wn + (long)o * HID_;
    float acc = bn[o];
    for (int c = 0; c < HID_; ++c) acc += x[c] * w[c];
    outn[i] = mish_f(acc);
}

// ---------------- kernel 5: edge out = mish((msg*k) @ We^T + be) (WMMA) -
// Same blocking as kernel 2; message scaling is applied while staging the
// A tile into LDS (per (row,head) thread -> single msg load per 32 elems).
__global__ __launch_bounds__(128) void edgeout_wmma_k(const _Float16* __restrict__ k16,
        const float* __restrict__ msg, const _Float16* __restrict__ We16,
        const float* __restrict__ be, float* __restrict__ oute) {
    __shared__ _Float16 As[16 * LDSP];
    const int tid = threadIdx.x;
    const long r0 = (long)blockIdx.x * 16;

    {   // stage: thread -> (row = tid>>3, head h = tid&7)
        int row = tid >> 3;
        int h   = tid & 7;
        long rr = r0 + row;
        int b   = (int)(rr >> 14);
        int rem = (int)(rr & 16383);                   // m*128 + n
        _Float16 mh = (_Float16)msg[(long)(b * NH_ + h) * (M_ * M_) + rem];
        const _Float16* src = k16 + rr * HID_ + h * D_;
        _Float16* dst = As + row * LDSP + h * D_;
        #pragma unroll
        for (int i = 0; i < D_; ++i) dst[i] = src[i] * mh;
    }
    __syncthreads();

    const int lane = tid & 31;
    const int wv   = tid >> 5;
    const int lrow = lane & 15;
    const int kb   = (lane < 16) ? 0 : 16;
    const int boff = (lane < 16) ? 0 : 16;

    v16h A[8];
    #pragma unroll
    for (int ks = 0; ks < 8; ++ks) {
        const _Float16* p = As + lrow * LDSP + ks * 32 + kb;
        v8h lo = *reinterpret_cast<const v8h*>(p);
        v8h hi = *reinterpret_cast<const v8h*>(p + 8);
        A[ks] = __builtin_shufflevector(lo, hi, 0,1,2,3,4,5,6,7,8,9,10,11,12,13,14,15);
    }

    v8f z; for (int i = 0; i < 8; ++i) z[i] = 0.f;
    v8f acc[4]; for (int t = 0; t < 4; ++t) acc[t] = z;

    #pragma unroll
    for (int ks = 0; ks < 8; ++ks) {
        #pragma unroll
        for (int t = 0; t < 4; ++t) {
            int o = 64 * wv + 16 * t + lrow;
            v16h bf = *reinterpret_cast<const v16h*>(We16 + (long)o * HID_ + ks * 32 + boff);
            acc[t] = __builtin_amdgcn_wmma_f32_16x16x32_f16(false, A[ks], false, bf,
                                                            (short)0, acc[t], false, false);
        }
    }

    const int rowhi = (lane < 16) ? 0 : 8;
    #pragma unroll
    for (int t = 0; t < 4; ++t) {
        int o = 64 * wv + 16 * t + lrow;
        float bias = be[o];
        #pragma unroll
        for (int g = 0; g < 8; ++g) {
            long row = r0 + g + rowhi;
            oute[row * HID_ + o] = mish_f(acc[t][g] + bias);
        }
    }
}

// -----------------------------------------------------------------------
extern "C" void kernel_launch(void* const* d_in, const int* in_sizes, int n_in,
                              void* d_out, int out_size, void* d_ws, size_t ws_size,
                              hipStream_t stream) {
    const float* node = (const float*)d_in[0];
    const float* edge = (const float*)d_in[1];
    const float* dist = (const float*)d_in[2];
    const float* mask = (const float*)d_in[3];
    const float* lam  = (const float*)d_in[4];
    const float* Wq = (const float*)d_in[5];  const float* bq = (const float*)d_in[6];
    const float* Wk = (const float*)d_in[7];  const float* bk = (const float*)d_in[8];
    const float* Wv = (const float*)d_in[9];  const float* bv = (const float*)d_in[10];
    const float* Wn = (const float*)d_in[11]; const float* bn = (const float*)d_in[12];
    const float* We = (const float*)d_in[13]; const float* be = (const float*)d_in[14];

    char* ws = (char*)d_ws;                   // ~77 MB used
    float*    q      = (float*)(ws + (0l  << 20));
    float*    v      = (float*)(ws + (1l  << 20));
    float*    nh     = (float*)(ws + (2l  << 20));
    _Float16* Wk16   = (_Float16*)(ws + (3l << 20));
    _Float16* We16   = (_Float16*)(ws + (3l << 20) + (1l << 18));
    float*    s      = (float*)(ws + (4l  << 20));   // 4 MB scores
    float*    msg    = (float*)(ws + (8l  << 20));   // 4 MB message
    float*    rowmax = (float*)(ws + (12l << 20));
    float*    rowsum = rowmax + B_ * NH_ * M_;
    float*    colmax = rowsum + B_ * NH_ * M_;
    float*    colsum = colmax + B_ * NH_ * M_;
    _Float16* k16    = (_Float16*)(ws + (13l << 20)); // 64 MB k (f16)

    float* outn = (float*)d_out;
    float* oute = outn + B_ * M_ * HID_;

    cvt_weights_k<<<512, 256, 0, stream>>>(Wk, We, Wk16, We16);
    qv_proj_k   <<<2048, 256, 0, stream>>>(node, Wq, bq, Wv, bv, q, v);
    kproj_wmma_k<<<R_EDGE / 16, 128, 0, stream>>>(edge, Wk16, bk, k16);
    scores_k    <<<R_EDGE / 256, 256, 0, stream>>>(q, k16, s);
    stats_k     <<<(B_ * NH_ * M_) / 256, 256, 0, stream>>>(s, mask, rowmax, rowsum, colmax, colsum);
    message_k   <<<(B_ * NH_ * M_ * M_) / 256, 256, 0, stream>>>(s, dist, mask, lam,
                                                                 rowmax, rowsum, colmax, colsum, msg);
    nodehid_k   <<<(B_ * M_ * HID_) / 256, 256, 0, stream>>>(msg, v, nh);
    nodeout_k   <<<(B_ * M_ * HID_) / 256, 256, 0, stream>>>(nh, Wn, bn, outn);
    edgeout_wmma_k<<<R_EDGE / 16, 128, 0, stream>>>(k16, msg, We16, be, oute);
}
